// density_wfa_finetune_7739531067862
// MI455X (gfx1250) — matrix operand order repaired and checked by reading, not compiled, not run on hardware
//
#include <hip/hip_runtime.h>
#include <hip/hip_bf16.h>
#include <cstdint>

typedef __attribute__((ext_vector_type(16))) _Float16 v16h;
typedef __attribute__((ext_vector_type(8)))  float    v8f;

#define N_TOT   4096
#define D_IN    8
#define T_LEN   256
#define R_DIM   128
#define NSAMP   16           // samples per workgroup
#define KMIX    10
#define NHEAD   192          // 80 mu | 80 sig | 10 alpha | 22 pad
#define NWG     (N_TOT / NSAMP)   // 256 workgroups
#define LOG2PI_F 1.8378770664093453f

// ---------------------------------------------------------------------------
// WMMA helpers (CDNA5 wave32, V_WMMA_F32_16X16X32_F16)
// ---------------------------------------------------------------------------
__device__ __forceinline__ v8f wmma16(v16h a, v16h b, v8f c) {
  // 8 args: (neg_a, A, neg_b, B, c_mod, C, reuse_a, reuse_b)
  return __builtin_amdgcn_wmma_f32_16x16x32_f16(false, a, false, b, (short)0, c,
                                                false, false);
}

// Build A fragment (16x32, f16) from a row-major f32 activation tile
// act[16][128] at K-offset k0.  A layout (contiguous): lanes 0-15 hold
// K=0..15; lanes 16-31 hold K=16..31; row M = lane&15.
__device__ __forceinline__ v16h make_afrag(const float* act, int m, int b0, int k0) {
  v16h a;
#pragma unroll
  for (int e = 0; e < 16; ++e) {
    int K = 2 * b0 + e;
    a[e] = (_Float16)act[m * R_DIM + k0 + K];
  }
  return a;
}

// C(16 x 16) tile for column-tile jt of act[16][128] @ W[128][16*nj].
// W is pre-packed as B fragments: Wf[(c*nj + jt)*512 + lane*16 + e],
// with B layout: lane = ((K>=16)?16:0) | n ; elem e = K&15.
__device__ __forceinline__ v8f gemm128(const float* act, const _Float16* Wf,
                                       int nj, int jt, int lane) {
  int m = lane & 15, b0 = (lane >> 4) * 8;
  v8f acc = {};
#pragma unroll
  for (int c = 0; c < 4; ++c) {
    v16h a = make_afrag(act, m, b0, c * 32);
    v16h b = *(const v16h*)(Wf + ((size_t)(c * nj + jt) * 512) + (size_t)lane * 16);
    acc = wmma16(a, b, acc);
  }
  return acc;
}

// ---------------------------------------------------------------------------
// Kernel 1: tranA = softmax(A, axis=2), written as f16 WMMA B-fragments.
// Row k = i*128+d of Aflat[16384][128]; fragment chunk c=k>>5, K=k&31.
// ---------------------------------------------------------------------------
__global__ void wfa_prep_tranA(const float* __restrict__ A,
                               _Float16* __restrict__ Bbuf) {
  __shared__ float red[128];
  int k = blockIdx.x;            // 0..16383
  int j = threadIdx.x;           // 0..127
  float v = A[(size_t)k * 128 + j];
  red[j] = v;
  __syncthreads();
  for (int s = 64; s > 0; s >>= 1) {
    if (j < s) red[j] = fmaxf(red[j], red[j + s]);
    __syncthreads();
  }
  float mx = red[0];
  __syncthreads();
  float ev = __expf(v - mx);
  red[j] = ev;
  __syncthreads();
  for (int s = 64; s > 0; s >>= 1) {
    if (j < s) red[j] += red[j + s];
    __syncthreads();
  }
  float p = ev / red[0];
  int c = k >> 5, K = k & 31;
  int jt = j >> 4, n = j & 15;
  int lane = ((K >> 4) << 4) | n;
  int e = K & 15;
  Bbuf[(((size_t)c * 8 + jt) * 32 + lane) * 16 + e] = (_Float16)p;
}

// ---------------------------------------------------------------------------
// Kernel 2: pack W_enc2 [128x128], W_nade [128x128], heads [128x192]
// (mu|sig|alpha|pad) into f16 B fragments.  112 blocks x 256 threads.
// ---------------------------------------------------------------------------
__global__ void wfa_pack(const float* __restrict__ W_enc2,
                         const float* __restrict__ W_nade,
                         const float* __restrict__ W_mu,
                         const float* __restrict__ W_sig,
                         const float* __restrict__ W_alpha,
                         _Float16* __restrict__ FragE,
                         _Float16* __restrict__ FragN,
                         _Float16* __restrict__ FragH) {
  int f = blockIdx.x;
  int mode, lf, nj;
  _Float16* dst;
  if (f < 32)      { mode = 0; lf = f;      nj = 8;  dst = FragE; }
  else if (f < 64) { mode = 1; lf = f - 32; nj = 8;  dst = FragN; }
  else             { mode = 2; lf = f - 64; nj = 12; dst = FragH; }
  int c = lf / nj, jt = lf % nj;
#pragma unroll
  for (int q = 0; q < 2; ++q) {
    int idx  = threadIdx.x * 2 + q;     // 0..511
    int lane = idx >> 4, e = idx & 15;
    int n = lane & 15;
    int K = ((lane >> 4) << 4) | e;
    int k = c * 32 + K, col = jt * 16 + n;
    float v;
    if (mode == 0)      v = W_enc2[k * 128 + col];
    else if (mode == 1) v = W_nade[k * 128 + col];
    else v = (col < 80)  ? W_mu[k * 80 + col]
           : (col < 160) ? W_sig[k * 80 + (col - 80)]
           : (col < 170) ? W_alpha[k * 10 + (col - 160)]
                         : 0.0f;
    dst[(size_t)lf * 512 + idx] = (_Float16)v;
  }
}

// ---------------------------------------------------------------------------
// Kernel 3: fused recurrence.  One WG = 16 samples, 8 waves = 8 column tiles.
// ---------------------------------------------------------------------------
__global__ __launch_bounds__(256, 1) void wfa_main(
    const float* __restrict__ X,
    const float* __restrict__ W_enc1, const float* __restrict__ b_enc1,
    const float* __restrict__ b_enc2, const float* __restrict__ init_w,
    const float* __restrict__ b_nade,
    const float* __restrict__ b_mu, const float* __restrict__ b_sig,
    const float* __restrict__ b_alpha,
    const _Float16* __restrict__ Bbuf,
    const _Float16* __restrict__ FragE,
    const _Float16* __restrict__ FragN,
    const _Float16* __restrict__ FragH,
    float* __restrict__ res_out,          // d_out[0..4095]
    float* __restrict__ ss_partial) {     // [T_LEN * NWG]
  __shared__ __align__(32) float tmpbuf[2][NSAMP * R_DIM];  // state (double buf)
  __shared__ __align__(32) float eact[NSAMP * R_DIM];       // encoder output
  __shared__ __align__(32) float gact[NSAMP * R_DIM];       // h1 / nade hidden
  __shared__ __align__(32) float heads[NSAMP * NHEAD];      // mu|sig|alpha
  __shared__ float Wenc1s[D_IN * 128];
  __shared__ float b1s[128], b2s[128], bns[128], bhs[NHEAD];
  __shared__ float xprev[NSAMP * D_IN], xcur[NSAMP * D_IN];
  __shared__ float red[256];
  __shared__ float compb[KMIX * NSAMP];
  __shared__ float sm2[2];

  const int tid  = threadIdx.x;
  const int lane = tid & 31;
  const int wave = tid >> 5;
  const int wg   = blockIdx.x;

  // ---- one-time setup -----------------------------------------------------
  for (int idx = tid; idx < D_IN * 128; idx += 256) Wenc1s[idx] = W_enc1[idx];
  if (tid < 128) { b1s[tid] = b_enc1[tid]; b2s[tid] = b_enc2[tid]; bns[tid] = b_nade[tid]; }
  if (tid < NHEAD)
    bhs[tid] = (tid < 80) ? b_mu[tid]
             : (tid < 160) ? b_sig[tid - 80]
             : (tid < 170) ? b_alpha[tid - 160] : 0.0f;
  if (tid == 0) {
    float mx = -1e30f;
    for (int j = 0; j < 128; ++j) mx = fmaxf(mx, init_w[j]);
    float s = 0.0f;
    for (int j = 0; j < 128; ++j) s += __expf(init_w[j] - mx);
    sm2[0] = mx; sm2[1] = s;
  }
  __syncthreads();
  if (tid < 128) {
    float p = __expf(init_w[tid] - sm2[0]) / sm2[1];
    for (int m = 0; m < NSAMP; ++m) tmpbuf[0][m * R_DIM + tid] = p;  // t=0 state
  }
  float res_acc = 0.0f;  // meaningful for tid < 16
  __syncthreads();

  // ---- time loop ----------------------------------------------------------
  for (int t = 0; t < T_LEN; ++t) {
    if (tid < 128) {
      int m = tid >> 3, d = tid & 7;
      size_t base = ((size_t)(wg * NSAMP + m) * D_IN + d) * T_LEN;
      xcur[m * D_IN + d] = X[base + t];
      if (t > 0) xprev[m * D_IN + d] = X[base + t - 1];
    }
    __syncthreads();

    if (t > 0) {
      const float* tprev = tmpbuf[(t - 1) & 1];
      float*       tcur  = tmpbuf[t & 1];

      // encoder stage 1: h1 = tanh(xprev @ W_enc1 + b1)  -> gact
      {
        int m = tid >> 4, hb = (tid & 15) * 8;
#pragma unroll
        for (int q = 0; q < 8; ++q) {
          int h = hb + q;
          float s = b1s[h];
#pragma unroll
          for (int d = 0; d < D_IN; ++d) s += xprev[m * D_IN + d] * Wenc1s[d * 128 + h];
          gact[m * 128 + h] = tanhf(s);
        }
      }
      __syncthreads();

      // encoder stage 2 (WMMA): e = h1 @ W_enc2 + b2 -> eact
      {
        v8f c = gemm128(gact, FragE, 8, wave, lane);
        int col = wave * 16 + (lane & 15);
#pragma unroll
        for (int r = 0; r < 8; ++r) {
          int m2 = r + ((lane >> 4) << 3);
          eact[m2 * 128 + col] = c[r] + b2s[col];
        }
      }
      __syncthreads();

      // transition (WMMA): tmp_new[n,j] = sum_{i,d} tmp[n,i] e[n,d] tranA[i,d,j]
      // Khatri-Rao A-fragments built on the fly in f16 (packed muls);
      // B streamed from L2-resident Bbuf; two accumulators break the
      // WMMA D->C serial chain and spread A-register WAR reuse.
      {
        int m = lane & 15, b0 = (lane >> 4) * 8;
        v8f acc0 = {}, acc1 = {};
        for (int d0 = 0; d0 < 4; ++d0) {
          v16h efh;                        // e-fragment, converted once
#pragma unroll
          for (int e = 0; e < 16; ++e) {
            int K = 2 * b0 + e;
            efh[e] = (_Float16)eact[m * 128 + d0 * 32 + K];
          }
#pragma unroll 4
          for (int i = 0; i < 128; i += 2) {
            _Float16 s0 = (_Float16)tprev[m * 128 + i];
            _Float16 s1 = (_Float16)tprev[m * 128 + i + 1];
            const _Float16* bp0 =
                Bbuf + (((size_t)(i * 4 + d0) * 8 + wave) * 512) + (size_t)lane * 16;
            const _Float16* bp1 = bp0 + 16384;           // next i (stride 4*8*512)
            __builtin_prefetch((const void*)(bp0 + 32768), 0, 0);  // two i ahead
            v16h a0 = efh * s0;                          // 8x v_pk_mul_f16
            v16h a1 = efh * s1;
            v16h bf0 = *(const v16h*)bp0;
            v16h bf1 = *(const v16h*)bp1;
            acc0 = wmma16(a0, bf0, acc0);
            acc1 = wmma16(a1, bf1, acc1);
          }
        }
        int col = wave * 16 + (lane & 15);
#pragma unroll
        for (int r = 0; r < 8; ++r) {
          int m2 = r + ((lane >> 4) << 3);
          tcur[m2 * 128 + col] = acc0[r] + acc1[r];
        }
      }
      __syncthreads();
    }

    const float* tcur = tmpbuf[t & 1];

    // Frobenius-norm partial: deterministic per-(t,wg) slot (no atomics)
    {
      float s = 0.0f;
      int base = tid * 8;
#pragma unroll
      for (int q = 0; q < 8; ++q) { float v = tcur[base + q]; s += v * v; }
      red[tid] = s;
      __syncthreads();
      for (int st = 128; st > 0; st >>= 1) {
        if (tid < st) red[tid] += red[tid + st];
        __syncthreads();
      }
      if (tid == 0) ss_partial[t * NWG + wg] = red[0];
      __syncthreads();
    }

    // NADE hidden (WMMA): g = tanh(tmp @ W_nade + b) -> gact
    {
      v8f c = gemm128(tcur, FragN, 8, wave, lane);
      int col = wave * 16 + (lane & 15);
#pragma unroll
      for (int r = 0; r < 8; ++r) {
        int m2 = r + ((lane >> 4) << 3);
        gact[m2 * 128 + col] = tanhf(c[r] + bns[col]);
      }
    }
    __syncthreads();

    // heads (WMMA): [mu|sig|alpha] = g @ W_heads + b_heads
    for (int jt = wave; jt < 12; jt += 8) {
      v8f c = gemm128(gact, FragH, 12, jt, lane);
      int col = jt * 16 + (lane & 15);
#pragma unroll
      for (int r = 0; r < 8; ++r) {
        int m2 = r + ((lane >> 4) << 3);
        heads[m2 * NHEAD + col] = c[r] + bhs[col];
      }
    }
    __syncthreads();

    // MDN log-likelihood
    if (tid < KMIX * NSAMP) {  // 160 (n,k) pairs
      int k = tid >> 4, n = tid & 15;
      float a = 0.0f;
#pragma unroll
      for (int d = 0; d < D_IN; ++d) {
        float mu = heads[n * NHEAD + k * D_IN + d];
        float sl = heads[n * NHEAD + 80 + k * D_IN + d];  // log(sigma)
        float z  = (xcur[n * D_IN + d] - mu) * __expf(-sl);
        a += -0.5f * z * z - sl;
      }
      a -= 0.5f * (float)D_IN * LOG2PI_F;
      compb[k * NSAMP + n] = a;
    }
    __syncthreads();
    if (tid < NSAMP) {
      int n = tid;
      float m1 = -1e30f, m2 = -1e30f;
      float lg[KMIX], tv[KMIX];
#pragma unroll
      for (int k = 0; k < KMIX; ++k) {
        lg[k] = heads[n * NHEAD + 160 + k];
        tv[k] = lg[k] + compb[k * NSAMP + n];
        m1 = fmaxf(m1, tv[k]);
        m2 = fmaxf(m2, lg[k]);
      }
      float s1 = 0.0f, s2 = 0.0f;
#pragma unroll
      for (int k = 0; k < KMIX; ++k) { s1 += __expf(tv[k] - m1); s2 += __expf(lg[k] - m2); }
      // logsumexp(log_softmax(lg) + comp) = lse(lg+comp) - lse(lg)
      res_acc += (m1 + __logf(s1)) - (m2 + __logf(s2));
    }
    __syncthreads();
  }

  if (tid < NSAMP) res_out[wg * NSAMP + tid] = res_acc;
}

// ---------------------------------------------------------------------------
// Kernel 4: nrm = sum_t sqrt(sum_wg ss[t][wg]) -> d_out[4096]
// ---------------------------------------------------------------------------
__global__ void wfa_finish(const float* __restrict__ ss_partial,
                           float* __restrict__ out_nrm) {
  __shared__ float red[256];
  int t = threadIdx.x;  // 256 threads, one per timestep
  float s = 0.0f;
  for (int w = 0; w < NWG; ++w) s += ss_partial[t * NWG + w];
  red[t] = sqrtf(s);
  __syncthreads();
  for (int st = 128; st > 0; st >>= 1) {
    if (t < st) red[t] += red[t + st];
    __syncthreads();
  }
  if (t == 0) out_nrm[0] = red[0];
}

// ---------------------------------------------------------------------------
// Host launcher
// ---------------------------------------------------------------------------
extern "C" void kernel_launch(void* const* d_in, const int* in_sizes, int n_in,
                              void* d_out, int out_size, void* d_ws, size_t ws_size,
                              hipStream_t stream) {
  const float* X       = (const float*)d_in[0];
  const float* W_enc1  = (const float*)d_in[1];
  const float* b_enc1  = (const float*)d_in[2];
  const float* W_enc2  = (const float*)d_in[3];
  const float* b_enc2  = (const float*)d_in[4];
  const float* A       = (const float*)d_in[5];
  const float* init_w  = (const float*)d_in[6];
  const float* W_nade  = (const float*)d_in[7];
  const float* b_nade  = (const float*)d_in[8];
  const float* W_mu    = (const float*)d_in[9];
  const float* b_mu    = (const float*)d_in[10];
  const float* W_sig   = (const float*)d_in[11];
  const float* b_sig   = (const float*)d_in[12];
  const float* W_alpha = (const float*)d_in[13];
  const float* b_alpha = (const float*)d_in[14];
  float* out = (float*)d_out;  // [4096] res, [1] nrm

  // workspace layout
  char* ws = (char*)d_ws;
  _Float16* Bbuf  = (_Float16*)(ws);                        // 16384*128*2 = 4 MB
  _Float16* FragE = (_Float16*)(ws + 4194304);              // 32 KB
  _Float16* FragN = (_Float16*)(ws + 4194304 + 32768);      // 32 KB
  _Float16* FragH = (_Float16*)(ws + 4194304 + 65536);      // 48 KB
  float*    ss    = (float*)   (ws + 4194304 + 65536 + 49152);  // 256*256*4

  hipLaunchKernelGGL(wfa_prep_tranA, dim3(16384), dim3(128), 0, stream, A, Bbuf);
  hipLaunchKernelGGL(wfa_pack, dim3(112), dim3(256), 0, stream,
                     W_enc2, W_nade, W_mu, W_sig, W_alpha, FragE, FragN, FragH);
  hipLaunchKernelGGL(wfa_main, dim3(NWG), dim3(256), 0, stream,
                     X, W_enc1, b_enc1, b_enc2, init_w, b_nade,
                     b_mu, b_sig, b_alpha, Bbuf, FragE, FragN, FragH,
                     out, ss);
  hipLaunchKernelGGL(wfa_finish, dim3(1), dim3(256), 0, stream, ss, out + N_TOT);
}
